// FusedMLPSoftplus_69286412419486
// MI455X (gfx1250) — compile-verified, matrix operand order checked by reading, not run on hardware
//
#include <hip/hip_runtime.h>

typedef __bf16 bf16;
typedef __attribute__((ext_vector_type(16))) __bf16 bf16x16;
typedef __attribute__((ext_vector_type(8)))  __bf16 bf16x8;
typedef __attribute__((ext_vector_type(4)))  __bf16 bf16x4;
typedef __attribute__((ext_vector_type(8)))  float  f32x8;

static constexpr int BM = 128;   // block tile M
static constexpr int BN = 128;   // block tile N
static constexpr int BK = 32;    // k-step (one wmma_f32_16x16x32_bf16)
static constexpr int LDSS = BK + 8;  // padded LDS row stride (80B: conflict-free b128)

// ---------------- prep kernels ----------------

__global__ __launch_bounds__(256) void cvt_f32_to_bf16_x4(
    const float* __restrict__ in, bf16* __restrict__ out, int n4) {
  int i = blockIdx.x * blockDim.x + threadIdx.x;
  if (i < n4) {
    float4 v = ((const float4*)in)[i];
    bf16x4 o;
    o.x = (bf16)v.x; o.y = (bf16)v.y; o.z = (bf16)v.z; o.w = (bf16)v.w;
    ((bf16x4*)out)[i] = o;
  }
}

// w[K][N] f32 (row-major) -> wt[N][K] bf16 (row-major)
__global__ __launch_bounds__(256) void transpose_cvt_w(
    const float* __restrict__ w, bf16* __restrict__ wt, int K, int N) {
  int idx = blockIdx.x * blockDim.x + threadIdx.x;
  if (idx < K * N) {
    int k = idx / N;
    int n = idx - k * N;
    wt[(size_t)n * K + k] = (bf16)w[idx];
  }
}

// ---------------- fused GEMM(+softplus) ----------------
// C[M,N] = act(A[M,K] @ W[K,N]) ; W passed transposed as Wt[N,K], all bf16,
// f32 accumulation in WMMA. OutT = bf16 (hidden) or float (final).

__device__ __forceinline__ float softplus_f(float v) {
  // branch-free, stable: softplus(v) = max(v,0) + log(1 + exp(-|v|))
  // arg of exp is <= 0 (no overflow); for large |v| the log term -> 0.
  // __expf/__logf lower to v_exp_f32 / v_log_f32 (hardware TRANS ops).
  float e = __expf(-__builtin_fabsf(v));
  return __builtin_fmaxf(v, 0.0f) + __logf(1.0f + e);
}

template <bool ACT, typename OutT>
__global__ __launch_bounds__(256) void mlp_gemm_wmma(
    const bf16* __restrict__ A, const bf16* __restrict__ Wt,
    OutT* __restrict__ C, int M, int K, int N) {
  __shared__ bf16 ldsA[BM * LDSS];
  __shared__ bf16 ldsB[BN * LDSS];

  const int tid  = threadIdx.x;
  const int lane = tid & 31;
  const int wave = tid >> 5;       // 0..7
  const int waveM = wave >> 2;     // 0..1  -> 64-row strip
  const int waveN = wave & 3;      // 0..3  -> 32-col strip

  const int bM = blockIdx.y * BM;
  const int bN = blockIdx.x * BN;

  // staging: 256 threads cover a 128x32 bf16 tile with 2x 16B loads each
  const int sRow = tid >> 2;          // 0..63
  const int sCol = (tid & 3) * 8;     // 0,8,16,24

  // WMMA 16-bit operand lane mapping (ISA 7.12.2):
  // lanes 0-15: row, K {0..7} then {16..23}; lanes 16-31: same row, K {8..15} then {24..31}
  const int laneHalf = lane >> 4;     // 0/1 -> k offset 0/8
  const int laneIdx  = lane & 15;     // row within 16x16 tile

  f32x8 acc[4][2] = {};

  const bf16* aBase = A  + (size_t)bM * K;
  const bf16* bBase = Wt + (size_t)bN * K;

  const int kTiles = K / BK;

  // --- prologue: stage k-tile 0 ---
  uint4 ra0 = *(const uint4*)(aBase + (size_t)(sRow)      * K + sCol);
  uint4 ra1 = *(const uint4*)(aBase + (size_t)(sRow + 64) * K + sCol);
  uint4 rb0 = *(const uint4*)(bBase + (size_t)(sRow)      * K + sCol);
  uint4 rb1 = *(const uint4*)(bBase + (size_t)(sRow + 64) * K + sCol);
  *(uint4*)&ldsA[(sRow)      * LDSS + sCol] = ra0;
  *(uint4*)&ldsA[(sRow + 64) * LDSS + sCol] = ra1;
  *(uint4*)&ldsB[(sRow)      * LDSS + sCol] = rb0;
  *(uint4*)&ldsB[(sRow + 64) * LDSS + sCol] = rb1;
  __syncthreads();

  for (int kt = 0; kt < kTiles; ++kt) {
    const bool hasNext = (kt + 1) < kTiles;
    if (hasNext) {
      const int k0 = (kt + 1) * BK;
      ra0 = *(const uint4*)(aBase + (size_t)(sRow)      * K + k0 + sCol);
      ra1 = *(const uint4*)(aBase + (size_t)(sRow + 64) * K + k0 + sCol);
      rb0 = *(const uint4*)(bBase + (size_t)(sRow)      * K + k0 + sCol);
      rb1 = *(const uint4*)(bBase + (size_t)(sRow + 64) * K + k0 + sCol);
    }

    // fragment loads from LDS: two 16B reads per lane per fragment
    bf16x16 afrag[4];
    bf16x16 bfrag[2];
#pragma unroll
    for (int i = 0; i < 4; ++i) {
      const int m = waveM * 64 + i * 16 + laneIdx;
      const bf16* p = &ldsA[m * LDSS + laneHalf * 8];
      bf16x8 lo = *(const bf16x8*)(p);
      bf16x8 hi = *(const bf16x8*)(p + 16);
      afrag[i] = __builtin_shufflevector(lo, hi, 0,1,2,3,4,5,6,7,8,9,10,11,12,13,14,15);
    }
#pragma unroll
    for (int j = 0; j < 2; ++j) {
      const int n = waveN * 32 + j * 16 + laneIdx;
      const bf16* p = &ldsB[n * LDSS + laneHalf * 8];
      bf16x8 lo = *(const bf16x8*)(p);
      bf16x8 hi = *(const bf16x8*)(p + 16);
      bfrag[j] = __builtin_shufflevector(lo, hi, 0,1,2,3,4,5,6,7,8,9,10,11,12,13,14,15);
    }

#pragma unroll
    for (int i = 0; i < 4; ++i)
#pragma unroll
      for (int j = 0; j < 2; ++j)
        acc[i][j] = __builtin_amdgcn_wmma_f32_16x16x32_bf16(
            /*neg_a=*/false, afrag[i], /*neg_b=*/false, bfrag[j],
            /*c_mod=*/(short)0, acc[i][j], /*reuse_a=*/false, /*reuse_b=*/false);

    if (hasNext) {
      __syncthreads();  // everyone done reading current LDS tiles
      *(uint4*)&ldsA[(sRow)      * LDSS + sCol] = ra0;
      *(uint4*)&ldsA[(sRow + 64) * LDSS + sCol] = ra1;
      *(uint4*)&ldsB[(sRow)      * LDSS + sCol] = rb0;
      *(uint4*)&ldsB[(sRow + 64) * LDSS + sCol] = rb1;
      __syncthreads();
    }
  }

  // epilogue: C/D layout (ISA 7.12.2): VGPR e, lanes 0-15 -> M=e, lanes 16-31 -> M=8+e; N=lane%16
#pragma unroll
  for (int i = 0; i < 4; ++i) {
#pragma unroll
    for (int j = 0; j < 2; ++j) {
      const int r0 = bM + waveM * 64 + i * 16 + laneHalf * 8;
      const int c  = bN + waveN * 32 + j * 16 + laneIdx;
#pragma unroll
      for (int e = 0; e < 8; ++e) {
        float v = acc[i][j][e];
        if (ACT) v = softplus_f(v);
        C[(size_t)(r0 + e) * N + c] = (OutT)v;
      }
    }
  }
}

// ---------------- launcher ----------------

extern "C" void kernel_launch(void* const* d_in, const int* in_sizes, int n_in,
                              void* d_out, int out_size, void* d_ws, size_t ws_size,
                              hipStream_t stream) {
  const float* x  = (const float*)d_in[0];
  const float* w1 = (const float*)d_in[1];
  const float* w2 = (const float*)d_in[2];
  const float* w3 = (const float*)d_in[3];
  float* out = (float*)d_out;

  const int K1 = 256, N1 = 1024, N2 = 1024, N3 = 256;
  const int M = in_sizes[0] / K1;  // 131072

  // workspace layout (256B aligned slots)
  char* ws = (char*)d_ws;
  size_t off = 0;
  auto take = [&](size_t bytes) -> char* {
    char* p = ws + off;
    off = (off + bytes + 255) & ~(size_t)255;
    return p;
  };
  bf16* xb  = (bf16*)take((size_t)M  * K1 * sizeof(bf16));
  bf16* h1  = (bf16*)take((size_t)M  * N1 * sizeof(bf16));
  bf16* h2  = (bf16*)take((size_t)M  * N2 * sizeof(bf16));
  bf16* w1t = (bf16*)take((size_t)K1 * N1 * sizeof(bf16));
  bf16* w2t = (bf16*)take((size_t)N1 * N2 * sizeof(bf16));
  bf16* w3t = (bf16*)take((size_t)N2 * N3 * sizeof(bf16));
  (void)ws_size; (void)n_in; (void)out_size;

  // prep: X -> bf16; W -> bf16 transposed
  {
    int n4 = (M * K1) / 4;
    cvt_f32_to_bf16_x4<<<(n4 + 255) / 256, 256, 0, stream>>>(x, xb, n4);
  }
  {
    int n = K1 * N1;
    transpose_cvt_w<<<(n + 255) / 256, 256, 0, stream>>>(w1, w1t, K1, N1);
  }
  {
    int n = N1 * N2;
    transpose_cvt_w<<<(n + 255) / 256, 256, 0, stream>>>(w2, w2t, N1, N2);
  }
  {
    int n = N2 * N3;
    transpose_cvt_w<<<(n + 255) / 256, 256, 0, stream>>>(w3, w3t, N2, N3);
  }

  dim3 blk(256);
  // layer 1: h1 = softplus(x @ w1)
  mlp_gemm_wmma<true,  bf16><<<dim3(N1 / BN, M / BM), blk, 0, stream>>>(xb, w1t, h1, M, K1, N1);
  // layer 2: h2 = softplus(h1 @ w2)
  mlp_gemm_wmma<true,  bf16><<<dim3(N2 / BN, M / BM), blk, 0, stream>>>(h1, w2t, h2, M, N1, N2);
  // layer 3: out = h2 @ w3  (f32 output, no activation)
  mlp_gemm_wmma<false, float><<<dim3(N3 / BN, M / BM), blk, 0, stream>>>(h2, w3t, out, M, N2, N3);
}
